// UResNet_35081292873834
// MI455X (gfx1250) — matrix operand order repaired and checked by reading, not compiled, not run on hardware
//
#include <hip/hip_runtime.h>

// ---------------------------------------------------------------------------
// CDNA5 (gfx1250, wave32) sparse UNet. All gather-GEMMs run on
// V_WMMA_F32_16X16X32_F16 (f16 A/B, f32 accumulate). Activations stored f16.
// Weights pre-transposed to [K][Cout][CinPad] so B fragments are contiguous.
// Each wave owns TWO 16-row tiles so every B fragment feeds two WMMAs.
// ---------------------------------------------------------------------------

typedef _Float16 half_t;
typedef __attribute__((ext_vector_type(16))) _Float16     v16h;
typedef __attribute__((ext_vector_type(8)))  _Float16     v8h;
typedef __attribute__((ext_vector_type(8)))  float        v8f;
typedef __attribute__((ext_vector_type(4)))  unsigned int v4u;

#define MAX_CIN_PAD 128          // max Cin seen: 2*64 = 128 (post0 concat, level 3)
#define LDS_STRIDE  (MAX_CIN_PAD + 8)  // +16B skew -> 4-bank rotation per row
#define BN_MAXC     192          // max BN channel count is 128; headroom

// ---------------------------------- elementwise helpers --------------------

__global__ void k_f32_to_f16(const float* __restrict__ x, half_t* __restrict__ y, int n) {
  int i = blockIdx.x * blockDim.x + threadIdx.x;
  if (i < n) y[i] = (half_t)x[i];
}

// W[k][c][n] (f32) -> Wt[k][n][cpad] (f16, zero-padded to CinPad)
__global__ void k_cvtw(const float* __restrict__ w, half_t* __restrict__ wt,
                       int K, int Cin, int Cout, int CinPad) {
  int i = blockIdx.x * blockDim.x + threadIdx.x;
  int total = K * Cout * CinPad;
  if (i < total) {
    int c = i % CinPad;
    int t = i / CinPad;
    int n = t % Cout;
    int k = t / Cout;
    wt[i] = (c < Cin) ? (half_t)w[((size_t)k * Cin + c) * Cout + n] : (half_t)0;
  }
}

__global__ void k_zero_f32(float* __restrict__ p, int n) {
  int i = blockIdx.x * blockDim.x + threadIdx.x;
  if (i < n) p[i] = 0.0f;
}

__global__ void k_add_f16(const half_t* __restrict__ a, const half_t* __restrict__ b,
                          half_t* __restrict__ y, int n) {
  int i = blockIdx.x * blockDim.x + threadIdx.x;
  if (i < n) y[i] = (half_t)((float)a[i] + (float)b[i]);
}

__global__ void k_concat(const half_t* __restrict__ a, const half_t* __restrict__ b,
                         half_t* __restrict__ y, int N, int C) {
  int i = blockIdx.x * blockDim.x + threadIdx.x;
  int C2 = 2 * C;
  if (i < N * C2) {
    int r = i / C2, c = i % C2;
    y[i] = (c < C) ? a[(size_t)r * C + c] : b[(size_t)r * C + (c - C)];
  }
}

// deconv -> gather-conv: idx[i][k] = (up_offset[i]==k) ? up_coarse[i] : -1
__global__ void k_dec_idx(const int* __restrict__ upc, const int* __restrict__ upo,
                          int* __restrict__ out, int N) {
  int i = blockIdx.x * blockDim.x + threadIdx.x;
  if (i < N) {
    int o = upo[i], c = upc[i];
#pragma unroll
    for (int k = 0; k < 8; ++k) out[(size_t)i * 8 + k] = (k == o) ? c : -1;
  }
}

// ---------------------------------- batch-norm -----------------------------

__global__ void k_bn_stats(const half_t* __restrict__ x, float* __restrict__ sums,
                           int N, int C) {
  __shared__ float ls[2 * BN_MAXC];
  for (int i = threadIdx.x; i < 2 * C; i += blockDim.x) ls[i] = 0.0f;
  __syncthreads();
  size_t total = (size_t)N * C;
  for (size_t i = (size_t)blockIdx.x * blockDim.x + threadIdx.x; i < total;
       i += (size_t)gridDim.x * blockDim.x) {
    int c = (int)(i % C);
    float v = (float)x[i];
    atomicAdd(&ls[c], v);
    atomicAdd(&ls[C + c], v * v);
  }
  __syncthreads();
  for (int i = threadIdx.x; i < 2 * C; i += blockDim.x) atomicAdd(&sums[i], ls[i]);
}

__global__ void k_bn_finalize(const float* __restrict__ sums, const float* __restrict__ g,
                              const float* __restrict__ b, float* __restrict__ ss,
                              int N, int C) {
  int c = blockIdx.x * blockDim.x + threadIdx.x;
  if (c < C) {
    float mean = sums[c] / (float)N;
    float var  = sums[C + c] / (float)N - mean * mean;
    float sc   = g[c] * rsqrtf(var + 1.0e-4f);
    ss[c]      = sc;
    ss[C + c]  = b[c] - mean * sc;
  }
}

__global__ void k_bn_apply(const half_t* __restrict__ x, const float* __restrict__ ss,
                           half_t* __restrict__ y, int N, int C) {
  int i = blockIdx.x * blockDim.x + threadIdx.x;
  if (i < N * C) {
    int c = i % C;
    float v = (float)x[i] * ss[c] + ss[C + c];
    y[i] = (half_t)fmaxf(v, 0.0f);
  }
}

// ---------------------------------- WMMA gather-conv -----------------------
//
// y[Nout x Cout] = sum_k gather(x, idx[:,k]) @ W[k]   (idx<0 -> zero row)
// idx==nullptr => identity gather with K==1 (1x1 conv / nin shortcut).
// Wt layout: [K][Cout][CinPad] f16, zero-padded -> B fragments are 2x16B
// contiguous aligned loads; no bounds guards in the inner loop.
//
// One wave per 32-row output tile (two 16-row WMMA M-tiles sharing each B
// fragment); NT = Cout/16 accumulator tiles per M-tile.

template <int NT>
__global__ __launch_bounds__(128)
void k_gather_conv(const half_t* __restrict__ x, const int* __restrict__ idx,
                   const half_t* __restrict__ Wt, half_t* __restrict__ y,
                   int Nout, int Cin, int Cout, int K) {
  __shared__ __align__(16) half_t smem[4][32 * LDS_STRIDE];

  const int wave = threadIdx.x >> 5;
  const int lane = threadIdx.x & 31;
  const int tile = blockIdx.x * 4 + wave;
  const int row0 = tile << 5;            // 32 rows per wave
  const int m    = lane & 15;

  const int KC     = (Cin + 31) >> 5;    // 32-wide K chunks
  const int CinPad = KC << 5;
  half_t* my = &smem[wave][0];

  v8f acc0[NT], acc1[NT];
#pragma unroll
  for (int t = 0; t < NT; ++t) {
    acc0[t] = (v8f){0, 0, 0, 0, 0, 0, 0, 0};
    acc1[t] = (v8f){0, 0, 0, 0, 0, 0, 0, 0};
  }

  const int lo  = (lane < 16) ? 0 : 8;   // A-fragment K sub-base per half-wave
  const int kbB = (lane < 16) ? 0 : 16;  // unused; B fragment uses lo like A
  const size_t wkStride = (size_t)Cout * CinPad;

  for (int k = 0; k < K; ++k) {
    // ---- gather 32 rows into LDS: every lane owns one row ----
    {
      int r = row0 + lane;
      int src = -1;
      if (r < Nout) src = idx ? idx[(size_t)r * K + k] : r;
      half_t* dst = &my[lane * LDS_STRIDE];
      if ((Cin & 7) == 0) {
        // vector path: 16B chunks, zero-fill padding chunks in the same loop
        const v4u* s4 = (const v4u*)(x + (size_t)src * Cin);
        v4u* d4 = (v4u*)dst;
        const int nv = CinPad >> 3;
        const int nx = (src >= 0) ? (Cin >> 3) : 0;
        const v4u z = {0, 0, 0, 0};
        for (int c = 0; c < nv; ++c) d4[c] = (c < nx) ? s4[c] : z;
      } else {
        // stem path (Cin == 1)
        const half_t* xr = x + (size_t)src * Cin;
        for (int c = 0; c < CinPad; ++c)
          dst[c] = (src >= 0 && c < Cin) ? xr[c] : (half_t)0;
      }
    }
    // overlap: prefetch next offset's weight slice while this one computes
    if (k + 1 < K)
      __builtin_prefetch(Wt + (size_t)(k + 1) * wkStride + (size_t)lane * 64, 0, 1);
    __syncthreads();

    const half_t* Wk = Wt + (size_t)k * wkStride;
    for (int cc = 0; cc < KC; ++cc) {
      const int cin0 = cc << 5;
      // A fragments: lanes<16 K={0..7,16..23}, lanes>=16 K={8..15,24..31}
      const half_t* arow0 = &my[m * LDS_STRIDE + cin0 + lo];
      const half_t* arow1 = arow0 + 16 * LDS_STRIDE;
      v8h a00 = *(const v8h*)arow0;
      v8h a01 = *(const v8h*)(arow0 + 16);
      v8h a10 = *(const v8h*)arow1;
      v8h a11 = *(const v8h*)(arow1 + 16);
      v16h af0 = __builtin_shufflevector(a00, a01, 0, 1, 2, 3, 4, 5, 6, 7,
                                         8, 9, 10, 11, 12, 13, 14, 15);
      v16h af1 = __builtin_shufflevector(a10, a11, 0, 1, 2, 3, 4, 5, 6, 7,
                                         8, 9, 10, 11, 12, 13, 14, 15);
#pragma unroll
      for (int t = 0; t < NT; ++t) {
        const int n = (t << 4) + m;
        // B fragment: Wt[k][n][cin0+lo ..+8] and [cin0+lo+16 ..+8],
        // shared by both M-tiles
        const half_t* bp = Wk + (size_t)n * CinPad + cin0 + lo;
        v8h b0 = *(const v8h*)bp;
        v8h b1 = *(const v8h*)(bp + 16);
        v16h bfrag = __builtin_shufflevector(b0, b1, 0, 1, 2, 3, 4, 5, 6, 7,
                                             8, 9, 10, 11, 12, 13, 14, 15);
        acc0[t] = __builtin_amdgcn_wmma_f32_16x16x32_f16(
            false, af0, false, bfrag, (short)0, acc0[t], false, false);
        acc1[t] = __builtin_amdgcn_wmma_f32_16x16x32_f16(
            false, af1, false, bfrag, (short)0, acc1[t], false, false);
      }
    }
    __syncthreads();  // protect LDS tile before next-k overwrite
  }

  // ---- store D: VGPR j -> M = j + (lane<16?0:8), N = lane&15 ----
  const int rowoff = (lane < 16) ? 0 : 8;
#pragma unroll
  for (int t = 0; t < NT; ++t) {
#pragma unroll
    for (int j = 0; j < 8; ++j) {
      int rr0 = row0 + rowoff + j;
      int rr1 = rr0 + 16;
      int col = (t << 4) + m;
      if (rr0 < Nout) y[(size_t)rr0 * Cout + col] = (half_t)acc0[t][j];
      if (rr1 < Nout) y[(size_t)rr1 * Cout + col] = (half_t)acc1[t][j];
    }
  }
}

// ---------------------------------- head -----------------------------------

__global__ void k_final(const half_t* __restrict__ x, const float* __restrict__ Wl,
                        const float* __restrict__ bl, float* __restrict__ out, int N) {
  int i = blockIdx.x * blockDim.x + threadIdx.x;
  if (i < N) {
    float a0 = bl[0], a1 = bl[1], a2 = bl[2], a3 = bl[3], a4 = bl[4];
#pragma unroll
    for (int c = 0; c < 16; ++c) {
      float v = (float)x[(size_t)i * 16 + c];
      a0 += v * Wl[c * 5 + 0];
      a1 += v * Wl[c * 5 + 1];
      a2 += v * Wl[c * 5 + 2];
      a3 += v * Wl[c * 5 + 3];
      a4 += v * Wl[c * 5 + 4];
    }
    out[(size_t)i * 5 + 0] = a0;
    out[(size_t)i * 5 + 1] = a1;
    out[(size_t)i * 5 + 2] = a2;
    out[(size_t)i * 5 + 3] = a3;
    out[(size_t)i * 5 + 4] = a4;
  }
}

// ---------------------------------- host orchestration ---------------------

extern "C" void kernel_launch(void* const* d_in, const int* in_sizes, int n_in,
                              void* d_out, int out_size, void* d_ws, size_t ws_size,
                              hipStream_t stream) {
  (void)n_in; (void)out_size; (void)ws_size;
  const int L = 5;
  const int NPc[5] = {16, 32, 48, 64, 80};

  // ---- flattened-input index map (dict insertion order) ----
  int pi = 1;  // index 0 = features
  const int iStem = pi++, iFinG = pi++, iFinB = pi++, iLinW = pi++, iLinB = pi++;
  struct Blk { int bn1g, bn1b, c1, bn2g, bn2b, c2, nin; };
  auto takeBlk = [&](bool hasNin) {
    Blk b; b.bn1g = pi++; b.bn1b = pi++; b.c1 = pi++;
    b.bn2g = pi++; b.bn2b = pi++; b.c2 = pi++;
    b.nin = hasNin ? pi++ : -1; return b;
  };
  struct Lvl { Blk pre0, pre1; int dg, db, dW, ug, ub, uW; Blk post0, post1; };
  Lvl lv[L];
  for (int l = 0; l < L; ++l) {
    lv[l].pre0 = takeBlk(false);
    lv[l].pre1 = takeBlk(false);
    if (l < L - 1) {
      lv[l].dg = pi++; lv[l].db = pi++; lv[l].dW = pi++;
      lv[l].ug = pi++; lv[l].ub = pi++; lv[l].uW = pi++;
      lv[l].post0 = takeBlk(true);
      lv[l].post1 = takeBlk(false);
    }
  }
  int iNbr[5], iDmap[4], iUpc[4], iUpo[4];
  for (int l = 0; l < L; ++l)     iNbr[l]  = pi++;
  for (int l = 0; l < L - 1; ++l) iDmap[l] = pi++;
  for (int l = 0; l < L - 1; ++l) iUpc[l]  = pi++;
  for (int l = 0; l < L - 1; ++l) iUpo[l]  = pi++;

  int N[5];
  for (int l = 0; l < L; ++l) N[l] = in_sizes[iNbr[l]] / 27;

  auto F32 = [&](int i) { return (const float*)d_in[i]; };
  auto I32 = [&](int i) { return (const int*)d_in[i]; };

  // ---- bump allocator over workspace ----
  char* base = (char*)d_ws;
  size_t off = 0;
  auto alloc = [&](size_t bytes) -> void* {
    off = (off + 255) & ~(size_t)255;
    void* p = base + off;
    off += bytes;
    return p;
  };

  float* statSums = (float*)alloc(2 * BN_MAXC * sizeof(float));
  float* statSS   = (float*)alloc(2 * BN_MAXC * sizeof(float));

  // f32 [K][Cin][Cout] -> f16 [K][Cout][CinPad] (transpose + pad), per call
  auto cvtw = [&](int i, int K, int Cin, int Cout) -> half_t* {
    int CinPad = (((Cin + 31) >> 5) << 5);
    int total  = K * Cout * CinPad;
    half_t* p = (half_t*)alloc((size_t)total * sizeof(half_t));
    k_cvtw<<<(total + 255) / 256, 256, 0, stream>>>(F32(i), p, K, Cin, Cout, CinPad);
    return p;
  };

  half_t* wStem = cvtw(iStem, 27, 1, 16);
  struct BlkW { half_t *c1, *c2, *nin; };
  struct LvlW { BlkW pre0, pre1, post0, post1; half_t *dW, *uW; };
  LvlW lw[L];
  for (int l = 0; l < L; ++l) {
    int a = NPc[l];
    lw[l].pre0 = {cvtw(lv[l].pre0.c1, 27, a, a), cvtw(lv[l].pre0.c2, 27, a, a), nullptr};
    lw[l].pre1 = {cvtw(lv[l].pre1.c1, 27, a, a), cvtw(lv[l].pre1.c2, 27, a, a), nullptr};
    if (l < L - 1) {
      int b = NPc[l + 1];
      lw[l].dW = cvtw(lv[l].dW, 8, a, b);
      lw[l].uW = cvtw(lv[l].uW, 8, b, a);
      lw[l].post0 = {cvtw(lv[l].post0.c1, 27, 2 * a, a),
                     cvtw(lv[l].post0.c2, 27, a, a),
                     cvtw(lv[l].post0.nin, 1, 2 * a, a)};
      lw[l].post1 = {cvtw(lv[l].post1.c1, 27, a, a),
                     cvtw(lv[l].post1.c2, 27, a, a), nullptr};
    }
  }

  // ---- activation / temp buffers ----
  half_t *X[5], *X2[5], *T[5], *U[5], *V[5], *Q[5];
  int* DIDX[4];
  for (int l = 0; l < L; ++l) {
    int a = NPc[l];
    int maxc = (l < L - 1) ? 2 * a : a;
    size_t n = (size_t)N[l];
    X[l]  = (half_t*)alloc(n * a * sizeof(half_t));
    X2[l] = (half_t*)alloc(n * a * sizeof(half_t));
    T[l]  = (half_t*)alloc(n * maxc * sizeof(half_t));
    U[l]  = (half_t*)alloc(n * a * sizeof(half_t));
    V[l]  = (half_t*)alloc(n * a * sizeof(half_t));
    Q[l]  = (l < L - 1) ? (half_t*)alloc(n * maxc * sizeof(half_t)) : nullptr;
    if (l < L - 1) DIDX[l] = (int*)alloc(n * 8 * sizeof(int));
  }
  half_t* feat16 = (half_t*)alloc((size_t)N[0] * sizeof(half_t));

  // ---- launch helpers ----
  auto launch_conv = [&](const half_t* x, const int* idx, const half_t* W, half_t* y,
                         int Nout, int Cin, int Cout, int K) {
    int tiles  = (Nout + 31) >> 5;   // 32 rows per wave
    int blocks = (tiles + 3) >> 2;   // 4 waves per block
    switch (Cout >> 4) {
      case 1: k_gather_conv<1><<<blocks, 128, 0, stream>>>(x, idx, W, y, Nout, Cin, Cout, K); break;
      case 2: k_gather_conv<2><<<blocks, 128, 0, stream>>>(x, idx, W, y, Nout, Cin, Cout, K); break;
      case 3: k_gather_conv<3><<<blocks, 128, 0, stream>>>(x, idx, W, y, Nout, Cin, Cout, K); break;
      case 4: k_gather_conv<4><<<blocks, 128, 0, stream>>>(x, idx, W, y, Nout, Cin, Cout, K); break;
      default: k_gather_conv<5><<<blocks, 128, 0, stream>>>(x, idx, W, y, Nout, Cin, Cout, K); break;
    }
  };

  auto bn_relu = [&](const half_t* x, int N_, int C, int gi, int bi, half_t* y) {
    k_zero_f32<<<1, 256, 0, stream>>>(statSums, 2 * C);
    int total = N_ * C;
    int nb = (total + 255) / 256; if (nb > 512) nb = 512;
    k_bn_stats<<<nb, 256, 0, stream>>>(x, statSums, N_, C);
    k_bn_finalize<<<1, 256, 0, stream>>>(statSums, F32(gi), F32(bi), statSS, N_, C);
    k_bn_apply<<<(total + 255) / 256, 256, 0, stream>>>(x, statSS, y, N_, C);
  };

  auto add = [&](const half_t* a, const half_t* b, half_t* y, int n) {
    k_add_f16<<<(n + 255) / 256, 256, 0, stream>>>(a, b, y, n);
  };

  // residual block: in (N x a) -> out (N x b).
  auto resblock = [&](const half_t* in, int N_, int a, int b, const int* nbrIdx,
                      const Blk& pb, const BlkW& wb,
                      half_t* tbuf, half_t* c1buf, half_t* c2buf, half_t* scbuf,
                      half_t* out) {
    bn_relu(in, N_, a, pb.bn1g, pb.bn1b, tbuf);
    launch_conv(tbuf, nbrIdx, wb.c1, c1buf, N_, a, b, 27);
    bn_relu(c1buf, N_, b, pb.bn2g, pb.bn2b, tbuf);
    launch_conv(tbuf, nbrIdx, wb.c2, c2buf, N_, b, b, 27);
    const half_t* sc = in;
    if (wb.nin) {  // 1x1 shortcut conv via identity gather (idx==nullptr, K=1)
      launch_conv(in, nullptr, wb.nin, scbuf, N_, a, b, 1);
      sc = scbuf;
    }
    add(sc, c2buf, out, N_ * b);
  };

  // ---- stem ----
  k_f32_to_f16<<<(N[0] + 255) / 256, 256, 0, stream>>>(F32(0), feat16, N[0]);
  launch_conv(feat16, I32(iNbr[0]), wStem, X[0], N[0], 1, 16, 27);

  // ---- down pass ----
  for (int l = 0; l < L; ++l) {
    int a = NPc[l];
    resblock(X[l], N[l], a, a, I32(iNbr[l]), lv[l].pre0, lw[l].pre0,
             T[l], U[l], V[l], nullptr, U[l]);
    resblock(U[l], N[l], a, a, I32(iNbr[l]), lv[l].pre1, lw[l].pre1,
             T[l], V[l], X[l], nullptr, X2[l]);
    if (l < L - 1) {
      bn_relu(X2[l], N[l], a, lv[l].dg, lv[l].db, T[l]);
      launch_conv(T[l], I32(iDmap[l]), lw[l].dW, X[l + 1],
                  N[l + 1], a, NPc[l + 1], 8);
    }
  }

  // ---- up pass ----
  for (int l = L - 2; l >= 0; --l) {
    int a = NPc[l], b = NPc[l + 1];
    bn_relu(X2[l + 1], N[l + 1], b, lv[l].ug, lv[l].ub, T[l + 1]);
    k_dec_idx<<<(N[l] + 255) / 256, 256, 0, stream>>>(I32(iUpc[l]), I32(iUpo[l]),
                                                      DIDX[l], N[l]);
    launch_conv(T[l + 1], DIDX[l], lw[l].uW, U[l], N[l], b, a, 8);  // hup -> U
    k_concat<<<((N[l] * 2 * a) + 255) / 256, 256, 0, stream>>>(X2[l], U[l], T[l],
                                                               N[l], a);
    resblock(T[l], N[l], 2 * a, a, I32(iNbr[l]), lv[l].post0, lw[l].post0,
             Q[l], U[l], V[l], X[l], U[l]);
    resblock(U[l], N[l], a, a, I32(iNbr[l]), lv[l].post1, lw[l].post1,
             Q[l], V[l], X[l], nullptr, X2[l]);
  }

  // ---- head: final BN+ReLU then 16->5 linear (f32 out) ----
  bn_relu(X2[0], N[0], 16, iFinG, iFinB, T[0]);
  k_final<<<(N[0] + 255) / 256, 256, 0, stream>>>(T[0], F32(iLinW), F32(iLinB),
                                                  (float*)d_out, N[0]);
}
